// ReCoMIL_46231027974357
// MI455X (gfx1250) — matrix operand order repaired and checked by reading, compiled, not run on hardware
//
#include <hip/hip_runtime.h>
#include <hip/hip_bf16.h>
#include <math.h>
#include <stdint.h>

// ---------------- problem constants ----------------
#define N_PTS 32768
#define D_IN  768
#define DD    512
#define KP0   128
#define NROWS (N_PTS + 16)

// ---------------- GEMM tiling ----------------
#define BM 128
#define BN 64
#define BK 32

typedef __attribute__((ext_vector_type(16))) __bf16 v16bf;
typedef __attribute__((ext_vector_type(8)))  float  v8f;

// exact parameter types for __builtin_amdgcn_global_load_async_to_lds_b128
// (from the clang diagnostic: vector_size(16) int, AS1 source / AS3 dest)
typedef int v4i32 __attribute__((vector_size(16)));
typedef __attribute__((address_space(1))) v4i32 as1_v4i32;
typedef __attribute__((address_space(3))) v4i32 as3_v4i32;

enum { ACT_NONE = 0, ACT_LEAKY = 1, ACT_RELU = 2, ACT_TANH = 3, ACT_SIG = 4 };

__device__ inline __bf16 f2bf(float f) {
  unsigned int u = __float_as_uint(f);
  u += 0x7FFFu + ((u >> 16) & 1u);          // round-to-nearest-even
  unsigned short s = (unsigned short)(u >> 16);
  __bf16 r;
  __builtin_memcpy(&r, &s, sizeof(r));
  return r;
}

// ---- CDNA5 async global->LDS copy (ASYNCcnt path) ----
__device__ inline void async_b128(const void* g, void* l) {
#if __has_builtin(__builtin_amdgcn_global_load_async_to_lds_b128)
  __builtin_amdgcn_global_load_async_to_lds_b128(
      (as1_v4i32*)(uintptr_t)g, (as3_v4i32*)(uintptr_t)l, 0, 0);
#else
  unsigned lds = (unsigned)(uintptr_t)l;
  unsigned long long ga = (unsigned long long)(uintptr_t)g;
  asm volatile("global_load_async_to_lds_b128 %0, %1, off"
               :: "v"(lds), "v"(ga) : "memory");
#endif
}

#if __has_builtin(__builtin_amdgcn_s_wait_asynccnt)
#define WAIT_ASYNC(n) __builtin_amdgcn_s_wait_asynccnt(n)
#else
#define WAIT_ASYNC(n) asm volatile("s_wait_asynccnt %0" :: "i"(n) : "memory")
#endif

__device__ inline float wave_sum(float v) {
#pragma unroll
  for (int o = 16; o > 0; o >>= 1) v += __shfl_xor(v, o, 32);
  return v;
}
__device__ inline float wave_max(float v) {
#pragma unroll
  for (int o = 16; o > 0; o >>= 1) v = fmaxf(v, __shfl_xor(v, o, 32));
  return v;
}

// =====================================================================
// Generic WMMA GEMM:  Out[M,Nn] = epilogue( A[M,K] @ op(B) + bias )
//   transB=1: B is torch Linear weight [Nn,K]  (compute A @ B^T)
//   transB=0: B is [K,Nn]
// epilogue order: +bias -> act -> *mulsrc -> +resid
// Double-buffered async staging of f32 tiles into LDS; bf16 conversion
// at fragment build (co-executes with v_wmma).
// Requirements: K % 32 == 0; all out-of-range tile reads must land in
// mapped memory (guaranteed by the workspace layout); stores are guarded.
// =====================================================================
__global__ __launch_bounds__(256)
void gemm_wmma(const float* __restrict__ A, int lda,
               const float* __restrict__ B, int ldb, int transB,
               const float* __restrict__ bias,
               const float* __restrict__ resid,
               const float* __restrict__ mulsrc,
               float* __restrict__ Out, int ldo,
               int M, int Nn, int K, int act)
{
  __shared__ float Af[2][BM][BK + 4];   // row stride 144B (16B multiple)
  __shared__ float Bf[2][BN][BK + 4];   // stored transposed: [n][k]

  const int tid  = threadIdx.x;
  const int bm   = blockIdx.y * BM;
  const int bn   = blockIdx.x * BN;
  const int wid  = tid >> 5;
  const int lane = tid & 31;
  const int wm   = (wid & 3) * 32;   // wave row offset in block tile
  const int wn   = (wid >> 2) * 32;  // wave col offset in block tile
  const int l    = lane & 15;
  const int hi   = lane >> 4;

  v8f acc[2][2];
#pragma unroll
  for (int a = 0; a < 2; a++)
#pragma unroll
    for (int c = 0; c < 2; c++)
#pragma unroll
      for (int v = 0; v < 8; v++) acc[a][c][v] = 0.0f;

  const int nt = K / BK;

  // async-issue one K-tile (A always; B only when transB: contiguous rows)
  auto issue_tile = [&](int kt, int buf) {
    const float* Abase = A + (size_t)bm * lda + (size_t)kt * BK;
#pragma unroll
    for (int i = 0; i < 4; i++) {          // 4 wave-instructions for A (16KB)
      int id = tid + i * 256;
      int r = id >> 3, kq = (id & 7) << 2;
      async_b128(Abase + (size_t)r * lda + kq, &Af[buf][r][kq]);
    }
    if (transB) {
      const float* Bbase = B + (size_t)bn * ldb + (size_t)kt * BK;
#pragma unroll
      for (int i = 0; i < 2; i++) {        // 2 wave-instructions for B (8KB)
        int id = tid + i * 256;
        int n = id >> 3, kq = (id & 7) << 2;
        async_b128(Bbase + (size_t)n * ldb + kq, &Bf[buf][n][kq]);
      }
    }
  };

  issue_tile(0, 0);

  for (int kt = 0; kt < nt; kt++) {
    const int buf = kt & 1;
    if (kt + 1 < nt) issue_tile(kt + 1, buf ^ 1);
    if (!transB) {
      // transpose-stage B tile synchronously (only small ctx GEMM uses this)
      const int k0 = kt * BK;
      for (int t = tid; t < BK * BN; t += 256) {
        int k = t >> 6, n = t & 63;
        Bf[buf][n][k] = B[(size_t)(k0 + k) * ldb + (bn + n)];
      }
    }
    // wait for this tile's async transfers (in-order completion),
    // leaving the next tile's transfers in flight
    if (kt + 1 < nt) {
      if (transB) { WAIT_ASYNC(6); } else { WAIT_ASYNC(4); }
    } else {
      WAIT_ASYNC(0);
    }
    __syncthreads();

    // ---- fragments per ISA 7.12.2 (16-bit A 16x32, B 32x16, f32 C/D) ----
    v16bf afr[2], bfr[2];
#pragma unroll
    for (int tm = 0; tm < 2; tm++) {
      int row = wm + tm * 16 + l;
      const float* ap = &Af[buf][row][0];
#pragma unroll
      for (int i = 0; i < 8; i++) {
        afr[tm][i]     = f2bf(ap[hi * 8 + i]);
        afr[tm][i + 8] = f2bf(ap[16 + hi * 8 + i]);
      }
    }
#pragma unroll
    for (int tn = 0; tn < 2; tn++) {
      int col = wn + tn * 16 + l;
      const float* bp = &Bf[buf][col][0];
#pragma unroll
      for (int i = 0; i < 8; i++) {
        bfr[tn][i]     = f2bf(bp[hi * 8 + i]);
        bfr[tn][i + 8] = f2bf(bp[16 + hi * 8 + i]);
      }
    }
#pragma unroll
    for (int tm = 0; tm < 2; tm++)
#pragma unroll
      for (int tn = 0; tn < 2; tn++)
        acc[tm][tn] = __builtin_amdgcn_wmma_f32_16x16x32_bf16(
            false, afr[tm], false, bfr[tn], (short)0, acc[tm][tn], false, false);
    __syncthreads();
  }

  // ---- epilogue + store ----
#pragma unroll
  for (int tm = 0; tm < 2; tm++) {
#pragma unroll
    for (int tn = 0; tn < 2; tn++) {
      int n = bn + wn + tn * 16 + l;
#pragma unroll
      for (int v = 0; v < 8; v++) {
        int m = bm + wm + tm * 16 + hi * 8 + v;
        if (m < M && n < Nn) {
          float x = acc[tm][tn][v];
          if (bias) x += bias[n];
          if      (act == ACT_LEAKY) x = (x > 0.0f) ? x : 0.01f * x;
          else if (act == ACT_RELU)  x = fmaxf(x, 0.0f);
          else if (act == ACT_TANH)  x = tanhf(x);
          else if (act == ACT_SIG)   x = 1.0f / (1.0f + __expf(-x));
          if (mulsrc) x *= mulsrc[(size_t)m * ldo + n];
          if (resid)  x += resid[(size_t)m * ldo + n];
          Out[(size_t)m * ldo + n] = x;
        }
      }
    }
  }
}

// =====================================================================
// wave-per-row helpers (wave32)
// =====================================================================
__global__ __launch_bounds__(256)
void rowsumsq(const float* __restrict__ X, float* __restrict__ out, int rows, int D)
{
  int wid = threadIdx.x >> 5, lane = threadIdx.x & 31;
  int row = blockIdx.x * 8 + wid;
  if (row >= rows) return;
  float s = 0.0f;
  for (int d = lane; d < D; d += 32) {
    float v = X[(size_t)row * D + d];
    s += v * v;
  }
  s = wave_sum(s);
  if (lane == 0) out[row] = s;
}

// S holds dot = P@C^T; rewrite in place with softmax(-sqrt(d2)/denom)
__global__ __launch_bounds__(256)
void softmax_sim(float* __restrict__ S, const float* __restrict__ pp,
                 const float* __restrict__ cc, const float* __restrict__ scale,
                 int rows, int K)
{
  int wid = threadIdx.x >> 5, lane = threadIdx.x & 31;
  int row = blockIdx.x * 8 + wid;
  if (row >= rows) return;
  float denom = fmaxf(1e-6f, scale[0]);
  int nl = K >> 5;              // K in {128,64,32}
  float loc[4];
  float mx = -1e30f;
  float prow = pp[row];
  for (int j = 0; j < nl; j++) {
    int k = lane + j * 32;
    float dot = S[(size_t)row * K + k];
    float d2  = fmaxf(prow + cc[k] - 2.0f * dot, 1e-12f);
    float x   = -sqrtf(d2) / denom;
    loc[j] = x;
    mx = fmaxf(mx, x);
  }
  mx = wave_max(mx);
  float s = 0.0f;
  for (int j = 0; j < nl; j++) { loc[j] = __expf(loc[j] - mx); s += loc[j]; }
  s = wave_sum(s);
  float inv = 1.0f / s;
  for (int j = 0; j < nl; j++) S[(size_t)row * K + lane + j * 32] = loc[j] * inv;
}

// X = LN(X + R) * g + b   (R may be null); D == 512
__global__ __launch_bounds__(256)
void ln_rows(float* __restrict__ X, const float* __restrict__ R,
             const float* __restrict__ g, const float* __restrict__ b, int rows)
{
  int wid = threadIdx.x >> 5, lane = threadIdx.x & 31;
  int row = blockIdx.x * 8 + wid;
  if (row >= rows) return;
  float loc[16];
  float s = 0.0f;
#pragma unroll
  for (int j = 0; j < 16; j++) {
    int d = lane + j * 32;
    float y = X[(size_t)row * DD + d];
    if (R) y += R[(size_t)row * DD + d];
    loc[j] = y;
    s += y;
  }
  float mean = wave_sum(s) * (1.0f / DD);
  float v = 0.0f;
#pragma unroll
  for (int j = 0; j < 16; j++) { float t = loc[j] - mean; v += t * t; }
  v = wave_sum(v) * (1.0f / DD);
  float inv = rsqrtf(v + 1e-5f);
#pragma unroll
  for (int j = 0; j < 16; j++) {
    int d = lane + j * 32;
    X[(size_t)row * DD + d] = (loc[j] - mean) * inv * g[d] + b[d];
  }
}

// ProtoReducer stage 1: H1[d,k] = relu( sum_j C[j,d]*w1[k,j] + b1[k] )
__global__ void reducer1(const float* __restrict__ C, const float* __restrict__ w1,
                         const float* __restrict__ b1, float* __restrict__ H1, int Kin)
{
  int idx = blockIdx.x * blockDim.x + threadIdx.x;
  if (idx >= DD * Kin) return;
  int d = idx / Kin, k = idx % Kin;
  float s = b1[k];
  for (int j = 0; j < Kin; j++) s += C[(size_t)j * DD + d] * w1[(size_t)k * Kin + j];
  H1[(size_t)d * Kin + k] = fmaxf(s, 0.0f);
}

// ProtoReducer stage 2 (writes transposed): Cn[ko,d] = sum_j H1[d,j]*w2[ko,j] + b2[ko]
__global__ void reducer2(const float* __restrict__ H1, const float* __restrict__ w2,
                         const float* __restrict__ b2, float* __restrict__ Cn,
                         int Kin, int Kout)
{
  int idx = blockIdx.x * blockDim.x + threadIdx.x;
  if (idx >= DD * Kout) return;
  int d = idx / Kout, ko = idx % Kout;
  float s = b2[ko];
  for (int j = 0; j < Kin; j++) s += H1[(size_t)d * Kin + j] * w2[(size_t)ko * Kin + j];
  Cn[(size_t)ko * DD + d] = s;
}

// scores[m] = dot(X[m,:], w) + b0
__global__ __launch_bounds__(256)
void rowdot(const float* __restrict__ X, const float* __restrict__ w,
            const float* __restrict__ b, float* __restrict__ out, int rows)
{
  int wid = threadIdx.x >> 5, lane = threadIdx.x & 31;
  int row = blockIdx.x * 8 + wid;
  if (row >= rows) return;
  float s = 0.0f;
  for (int d = lane; d < DD; d += 32) s += X[(size_t)row * DD + d] * w[d];
  s = wave_sum(s);
  if (lane == 0) out[row] = s + b[0];
}

__global__ __launch_bounds__(1024)
void softmax_stats(const float* __restrict__ sc, int n, float* __restrict__ stats)
{
  __shared__ float red[1024];
  int tid = threadIdx.x;
  float mx = -1e30f;
  for (int i = tid; i < n; i += 1024) mx = fmaxf(mx, sc[i]);
  red[tid] = mx; __syncthreads();
  for (int o = 512; o > 0; o >>= 1) {
    if (tid < o) red[tid] = fmaxf(red[tid], red[tid + o]);
    __syncthreads();
  }
  mx = red[0]; __syncthreads();
  float s = 0.0f;
  for (int i = tid; i < n; i += 1024) s += __expf(sc[i] - mx);
  red[tid] = s; __syncthreads();
  for (int o = 512; o > 0; o >>= 1) {
    if (tid < o) red[tid] += red[tid + o];
    __syncthreads();
  }
  if (tid == 0) { stats[0] = mx; stats[1] = red[0]; }
}

// bag[d] = (1/sum) * sum_m exp(sc[m]-mx) * Z[m,d]   (one block per column)
__global__ __launch_bounds__(256)
void wcolsum(const float* __restrict__ Z, const float* __restrict__ sc,
             const float* __restrict__ stats, float* __restrict__ bag, int rows)
{
  __shared__ float red[256];
  int d = blockIdx.x;
  float mx = stats[0], inv = 1.0f / stats[1];
  float s = 0.0f;
  for (int m = threadIdx.x; m < rows; m += 256)
    s += __expf(sc[m] - mx) * Z[(size_t)m * DD + d];
  red[threadIdx.x] = s; __syncthreads();
  for (int o = 128; o > 0; o >>= 1) {
    if (threadIdx.x < o) red[threadIdx.x] += red[threadIdx.x + o];
    __syncthreads();
  }
  if (threadIdx.x == 0) bag[d] = red[0] * inv;
}

// h = relu(bag @ W_h^T + b_h); logits = h @ W_c^T + b_c
__global__ __launch_bounds__(512)
void head_kernel(const float* __restrict__ bag, const float* __restrict__ Wh,
                 const float* __restrict__ bh, const float* __restrict__ Wc,
                 const float* __restrict__ bc, float* __restrict__ out)
{
  __shared__ float h[DD];
  int j = threadIdx.x;
  float s = bh[j];
  for (int d = 0; d < DD; d++) s += bag[d] * Wh[(size_t)j * DD + d];
  h[j] = fmaxf(s, 0.0f);
  __syncthreads();
  if (j < 2) {
    float t = bc[j];
    for (int k = 0; k < DD; k++) t += h[k] * Wc[(size_t)j * DD + k];
    out[j] = t;
  }
}

// =====================================================================
// host side
// =====================================================================
static inline void launch_gemm(hipStream_t s,
                               const float* A, int lda,
                               const float* B, int ldb, int transB,
                               const float* bias, const float* resid, const float* mul,
                               float* Out, int ldo, int M, int Nn, int K, int act)
{
  dim3 grid((Nn + BN - 1) / BN, (M + BM - 1) / BM), blk(256);
  gemm_wmma<<<grid, blk, 0, s>>>(A, lda, B, ldb, transB, bias, resid, mul,
                                 Out, ldo, M, Nn, K, act);
}

extern "C" void kernel_launch(void* const* d_in, const int* in_sizes, int n_in,
                              void* d_out, int out_size, void* d_ws, size_t ws_size,
                              hipStream_t stream)
{
  (void)in_sizes; (void)n_in; (void)out_size; (void)ws_size;
  const float* data   = (const float*)d_in[0];
  const float* protos = (const float*)d_in[1];
  const float* W_pp   = (const float*)d_in[2];
  const float* b_pp   = (const float*)d_in[3];
  const float* W_cp   = (const float*)d_in[4];
  const float* b_cp   = (const float*)d_in[5];
  const float* scal   = (const float*)d_in[6];
  const float* enh_w1 = (const float*)d_in[7];
  const float* enh_b1 = (const float*)d_in[8];
  const float* enh_w2 = (const float*)d_in[9];
  const float* enh_b2 = (const float*)d_in[10];
  const float* ln_g   = (const float*)d_in[11];
  const float* ln_b   = (const float*)d_in[12];
  const float* red_w1[3] = {(const float*)d_in[13], (const float*)d_in[17], (const float*)d_in[21]};
  const float* red_b1[3] = {(const float*)d_in[14], (const float*)d_in[18], (const float*)d_in[22]};
  const float* red_w2[3] = {(const float*)d_in[15], (const float*)d_in[19], (const float*)d_in[23]};
  const float* red_b2[3] = {(const float*)d_in[16], (const float*)d_in[20], (const float*)d_in[24]};
  const float* W_proc = (const float*)d_in[25];
  const float* b_proc = (const float*)d_in[26];
  const float* g_an   = (const float*)d_in[27];
  const float* b_an   = (const float*)d_in[28];
  const float* W_t    = (const float*)d_in[29];
  const float* b_t    = (const float*)d_in[30];
  const float* W_g    = (const float*)d_in[31];
  const float* b_g    = (const float*)d_in[32];
  const float* W_s    = (const float*)d_in[33];
  const float* b_s    = (const float*)d_in[34];
  const float* W_h    = (const float*)d_in[35];
  const float* b_h    = (const float*)d_in[36];
  const float* W_c    = (const float*)d_in[37];
  const float* b_c    = (const float*)d_in[38];

  // ---- workspace carve-up ----
  // NOTE: B0,B1,B2,S are adjacent so over-tile A reads (rows >= M in the
  // Zcat GEMMs) land in mapped workspace; results for those rows are never
  // stored (epilogue guards).
  size_t off = 0;
  auto alloc = [&](size_t nf) {
    float* p = (float*)((char*)d_ws + off);
    off += nf * sizeof(float);
    return p;
  };
  float* B0    = alloc((size_t)NROWS * DD);   // P, then Zcat
  float* B1    = alloc((size_t)NROWS * DD);   // H / ctx / Z
  float* B2    = alloc((size_t)NROWS * DD);   // t, then t*g
  float* S     = alloc((size_t)N_PTS * KP0);  // dot / routing A
  float* Cb0   = alloc((size_t)KP0 * DD);
  float* Cb1   = alloc((size_t)KP0 * DD);
  float* RH    = alloc((size_t)DD * KP0);     // reducer hidden
  float* pp    = alloc(N_PTS);
  float* cc    = alloc(KP0);
  float* sc    = alloc(NROWS);
  float* stats = alloc(8);
  float* bag   = alloc(DD);

  const int KS[4] = {128, 64, 32, 16};
  const int rowblk = (N_PTS + 7) / 8;

  // P = leaky(data @ W_pp^T + b_pp) ; C = leaky(protos @ W_cp^T + b_cp)
  launch_gemm(stream, data, D_IN, W_pp, D_IN, 1, b_pp, nullptr, nullptr,
              B0, DD, N_PTS, DD, D_IN, ACT_LEAKY);
  launch_gemm(stream, protos, D_IN, W_cp, D_IN, 1, b_cp, nullptr, nullptr,
              Cb0, DD, KP0, DD, D_IN, ACT_LEAKY);

  float* Ccur = Cb0;
  float* Cnxt = Cb1;
  for (int i = 0; i < 3; i++) {
    int Kin = KS[i], Kout = KS[i + 1];
    rowsumsq<<<rowblk, 256, 0, stream>>>(B0, pp, N_PTS, DD);
    rowsumsq<<<(Kin + 7) / 8, 256, 0, stream>>>(Ccur, cc, Kin, DD);
    // dot = P @ C^T  (Linear form with W = C)
    launch_gemm(stream, B0, DD, Ccur, DD, 1, nullptr, nullptr, nullptr,
                S, Kin, N_PTS, Kin, DD, ACT_NONE);
    softmax_sim<<<rowblk, 256, 0, stream>>>(S, pp, cc, scal, N_PTS, Kin);
    // ctx = A @ C
    launch_gemm(stream, S, Kin, Ccur, DD, 0, nullptr, nullptr, nullptr,
                B1, DD, N_PTS, DD, Kin, ACT_NONE);
    // P = LN(P + ctx)
    ln_rows<<<rowblk, 256, 0, stream>>>(B0, B1, ln_g + (size_t)i * DD,
                                        ln_b + (size_t)i * DD, N_PTS);
    // P = P + lin2(relu(lin1(P)))
    launch_gemm(stream, B0, DD, enh_w1 + (size_t)i * DD * DD, DD, 1,
                enh_b1 + (size_t)i * DD, nullptr, nullptr,
                B1, DD, N_PTS, DD, DD, ACT_RELU);
    launch_gemm(stream, B1, DD, enh_w2 + (size_t)i * DD * DD, DD, 1,
                enh_b2 + (size_t)i * DD, B0, nullptr,
                B0, DD, N_PTS, DD, DD, ACT_NONE);
    // ProtoReducer along K axis
    reducer1<<<(DD * Kin + 255) / 256, 256, 0, stream>>>(Ccur, red_w1[i], red_b1[i], RH, Kin);
    reducer2<<<(DD * Kout + 255) / 256, 256, 0, stream>>>(RH, red_w2[i], red_b2[i], Cnxt, Kin, Kout);
    float* t = Ccur; Ccur = Cnxt; Cnxt = t;
  }

  // Zcat = [P ; C_final(16xD)]
  (void)hipMemcpyAsync(B0 + (size_t)N_PTS * DD, Ccur,
                       (size_t)16 * DD * sizeof(float),
                       hipMemcpyDeviceToDevice, stream);

  const int zrowblk = (NROWS + 7) / 8;
  // Z = relu(Zcat @ W_proc^T + b_proc); Z = LN(Z)
  launch_gemm(stream, B0, DD, W_proc, DD, 1, b_proc, nullptr, nullptr,
              B1, DD, NROWS, DD, DD, ACT_RELU);
  ln_rows<<<zrowblk, 256, 0, stream>>>(B1, nullptr, g_an, b_an, NROWS);
  // t = tanh(Z W_t^T + b_t); tg = sigmoid(Z W_g^T + b_g) * t
  launch_gemm(stream, B1, DD, W_t, DD, 1, b_t, nullptr, nullptr,
              B2, DD, NROWS, DD, DD, ACT_TANH);
  launch_gemm(stream, B1, DD, W_g, DD, 1, b_g, nullptr, B2,
              B2, DD, NROWS, DD, DD, ACT_SIG);
  // scores, softmax pooling, head
  rowdot<<<zrowblk, 256, 0, stream>>>(B2, W_s, b_s, sc, NROWS);
  softmax_stats<<<1, 1024, 0, stream>>>(sc, NROWS, stats);
  wcolsum<<<DD, 256, 0, stream>>>(B1, sc, stats, bag, NROWS);
  head_kernel<<<1, 512, 0, stream>>>(bag, W_h, b_h, W_c, b_c, (float*)d_out);
}
